// SelfAttention_67577015435858
// MI455X (gfx1250) — matrix operand order repaired and checked by reading, not compiled, not run on hardware
//
#include <hip/hip_runtime.h>

// Self-attention forward, B=16, T=2048, D_MODEL=512, D_HEAD=64, fp32 I/O.
// bf16 WMMA (v_wmma_f32_16x16x32_bf16) everywhere, f32 accumulation, wave32
// flash attention. TDM tensor_load_to_lds stages the x tile; global_prefetch
// warms next key tiles. QKV loop restructured (k outer, QKV fused per step)
// to cut register pressure and amortize s_wait_loadcnt over 3 WMMAs.

typedef __bf16 bf16;
typedef __attribute__((ext_vector_type(16))) __bf16 v16bf;
typedef __attribute__((ext_vector_type(8)))  __bf16 v8bf;
typedef __attribute__((ext_vector_type(8)))  float  v8f;
typedef __attribute__((ext_vector_type(4)))  unsigned int v4u;
typedef __attribute__((ext_vector_type(8)))  int v8i;
typedef __attribute__((ext_vector_type(4)))  int v4i;

static constexpr int kB = 16;
static constexpr int kT = 2048;
static constexpr int kD = 512;   // d_model
static constexpr int kH = 64;    // d_head

#if defined(__has_builtin)
#if __has_builtin(__builtin_amdgcn_tensor_load_to_lds) && __has_builtin(__builtin_amdgcn_s_wait_tensorcnt)
#define HAVE_TDM 1
#endif
#endif
#ifndef HAVE_TDM
#define HAVE_TDM 0
#endif

// ---------------------------------------------------------------------------
// Fragment loaders (wave32 WMMA, 16-bit operands; contiguous k order shared by A and B)
// A 16x32 (MxK): lane l -> row (l&15); 16 contiguous halves = K[ koff .. koff+15 ],
//                koff = 16*(l>>4).
// ---------------------------------------------------------------------------
static __device__ __forceinline__
v16bf load_a_frag(const bf16* base, int ld, int row0, int k0, int lane) {
  int r    = lane & 15;
  int koff = (lane >> 4) << 4;
  const bf16* p = base + (size_t)(row0 + r) * ld + k0 + koff;
  return *(const v16bf*)(p);
}

// B 32x16 (KxN) sourced from an N-major (row = N column) matrix:
// lane l -> column (l&15); 16 contiguous halves = K[ 16*(l>>4) .. +15 ].
static __device__ __forceinline__
v16bf load_b_frag(const bf16* base, int ld, int col0, int k0, int lane) {
  int n  = lane & 15;
  int kb = k0 + ((lane >> 4) << 4);
  const bf16* p = base + (size_t)(col0 + n) * ld + kb;
  return *(const v16bf*)(p);
}

static __device__ __forceinline__
v8f wmma_bf16(v16bf a, v16bf b, v8f c) {
  return __builtin_amdgcn_wmma_f32_16x16x32_bf16(
      false, a, false, b, (short)0, c, false, false);
}

#if HAVE_TDM
// TDM: async-load a 16x512 f32 tile (row stride 512 floats) into LDS.
// D# per ISA 08 §8: group0 = count=1 | lds_addr | global_addr | type=2;
// group1 = data_size=4B, tensor_dim0=512, tensor_dim1=16, tile 512x16,
// tensor_dim0_stride=512; no padding / multicast / iterate.
static __device__ __forceinline__
void tdm_load_tile_f32(const float* gsrc, unsigned lds_off) {
  unsigned long long ga = (unsigned long long)(uintptr_t)gsrc;
  v4u g0;
  g0.x = 1u;
  g0.y = lds_off;
  g0.z = (unsigned)(ga & 0xffffffffu);
  g0.w = (unsigned)((ga >> 32) & 0x01ffffffu) | (2u << 30);
  v8i g1;
  g1[0] = (int)(2u << 16);
  g1[1] = (int)((512u & 0xffffu) << 16);
  g1[2] = (int)(16u << 16);
  g1[3] = (int)(512u << 16);
  g1[4] = 16;
  g1[5] = 512;
  g1[6] = 0;
  g1[7] = 0;
  v4i z4 = {0, 0, 0, 0};
#if __clang_major__ >= 23
  v8i z8 = {0, 0, 0, 0, 0, 0, 0, 0};
  __builtin_amdgcn_tensor_load_to_lds(g0, g1, z4, z4, z8, 0);
#else
  __builtin_amdgcn_tensor_load_to_lds(g0, g1, z4, z4, 0);
#endif
}
#endif

// ---------------------------------------------------------------------------
// Kernel 0: convert + transpose weights -> bf16 Wt[3][64][512] (N-major)
// ---------------------------------------------------------------------------
__global__ void __launch_bounds__(256)
wt_transpose_kernel(const float* __restrict__ Wq, const float* __restrict__ Wk,
                    const float* __restrict__ Wv, bf16* __restrict__ Wt) {
  const float* W = (blockIdx.x == 0) ? Wq : (blockIdx.x == 1) ? Wk : Wv;
  bf16* dst = Wt + (size_t)blockIdx.x * kH * kD;
  for (int idx = threadIdx.x; idx < kD * kH; idx += blockDim.x) {
    int k = idx >> 6;
    int n = idx & (kH - 1);
    dst[(size_t)n * kD + k] = (bf16)W[idx];
  }
}

// ---------------------------------------------------------------------------
// Kernel 1: QKV projection. One block = one (batch, 16-row) x tile, staged
// via TDM then converted to bf16 in LDS. 4 waves; wave w owns head-dim
// N-tile w and accumulates Q, K and V simultaneously: per k-step one A
// fragment feeds three WMMAs (one wait per three matrix ops, ~80 VGPRs).
// ---------------------------------------------------------------------------
__global__ void __launch_bounds__(128)
qkv_proj_kernel(const float* __restrict__ x, const bf16* __restrict__ Wt,
                bf16* __restrict__ Qs, bf16* __restrict__ Kb,
                bf16* __restrict__ Vt) {
  __shared__ __align__(128) float xs_f32[16][kD];   // 32 KB TDM landing pad
  __shared__ __align__(128) bf16  xs[16][kD];       // 16 KB
  const int b = blockIdx.x >> 7;                    // T/16 = 128 tiles/batch
  const int R = (blockIdx.x & 127) << 4;
  const float* xrow = x + ((size_t)b * kT + R) * kD;

#if HAVE_TDM
  if (threadIdx.x < 32) {                           // wave 0 issues the DMA
    tdm_load_tile_f32(xrow, (unsigned)(uintptr_t)&xs_f32[0][0]);
    __builtin_amdgcn_s_wait_tensorcnt(0);
  }
  __syncthreads();
  for (int idx = threadIdx.x; idx < 16 * kD; idx += 128)
    xs[idx >> 9][idx & (kD - 1)] = (bf16)xs_f32[idx >> 9][idx & (kD - 1)];
#else
  for (int idx = threadIdx.x; idx < 16 * kD; idx += 128)
    xs[idx >> 9][idx & (kD - 1)] = (bf16)xrow[idx];
#endif
  __syncthreads();

  const int lane  = threadIdx.x & 31;
  const int wave  = threadIdx.x >> 5;
  const int rbase = (lane >> 4) << 3;
  const int nlo   = lane & 15;
  const int ncol  = wave * 16;                      // this wave's N-tile

  const bf16* wq = Wt;
  const bf16* wk = Wt + (size_t)1 * kH * kD;
  const bf16* wv = Wt + (size_t)2 * kH * kD;

  v8f accQ = {}, accK = {}, accV = {};
  #pragma unroll
  for (int kc = 0; kc < kD; kc += 32) {
    v16bf a  = load_a_frag(&xs[0][0], kD, 0, kc, lane);
    v16bf bq = load_b_frag(wq, kD, ncol, kc, lane);
    v16bf bk = load_b_frag(wk, kD, ncol, kc, lane);
    v16bf bv = load_b_frag(wv, kD, ncol, kc, lane);
    accQ = wmma_bf16(a, bq, accQ);
    accK = wmma_bf16(a, bk, accK);
    accV = wmma_bf16(a, bv, accV);
  }

  const int n = ncol + nlo;
  #pragma unroll
  for (int i = 0; i < 8; i++) {
    const size_t trow = (size_t)b * kT + R + rbase + i;
    Qs[trow * kH + n] = (bf16)(accQ[i] * 0.125f);   // fold 1/sqrt(d_head)
    Kb[trow * kH + n] = (bf16)accK[i];
    Vt[((size_t)b * kH + n) * kT + R + rbase + i] = (bf16)accV[i];
  }
}

// ---------------------------------------------------------------------------
// Kernel 2: causal flash attention. One wave = 16 query rows; 8 waves/block.
// Key step = 32 keys: 4 WMMAs for S, shuffle-based online softmax, P through
// per-wave LDS into an A fragment, 4 WMMAs for P*V. V fragments fetched
// early to overlap softmax; next step prefetched via global_prefetch_b8.
// ---------------------------------------------------------------------------
__global__ void __launch_bounds__(256)
attn_kernel(const bf16* __restrict__ Qs, const bf16* __restrict__ Kb,
            const bf16* __restrict__ Vt, float* __restrict__ out) {
  __shared__ bf16 pbuf[8][16][32];
  const int lane = threadIdx.x & 31;
  const int wave = threadIdx.x >> 5;
  const int tile = blockIdx.x * 8 + wave;
  const int b  = tile >> 7;
  const int q0 = (tile & 127) << 4;

  const bf16* Qb  = Qs + (size_t)b * kT * kH;
  const bf16* Kbb = Kb + (size_t)b * kT * kH;
  const bf16* Vtb = Vt + (size_t)b * kH * kT;

  const v16bf qf0 = load_a_frag(Qb, kH, q0, 0,  lane);
  const v16bf qf1 = load_a_frag(Qb, kH, q0, 32, lane);

  v8f o0 = {}, o1 = {}, o2 = {}, o3 = {};
  float m_i[8], l_i[8];
  #pragma unroll
  for (int i = 0; i < 8; i++) { m_i[i] = -__builtin_inff(); l_i[i] = 0.f; }

  const int rbase = (lane >> 4) << 3;
  const int nlo   = lane & 15;
  const int jend  = q0 + 16;                        // need keys <= q0+15

  for (int jb = 0; jb < jend; jb += 32) {
    // V fragments first: independent of S, overlap the softmax below.
    v16bf vf0 = load_b_frag(Vtb, kT, 0,  jb, lane);
    v16bf vf1 = load_b_frag(Vtb, kT, 16, jb, lane);
    v16bf vf2 = load_b_frag(Vtb, kT, 32, jb, lane);
    v16bf vf3 = load_b_frag(Vtb, kT, 48, jb, lane);

    // K fragments + S = Q*K^T (Q already carries 1/sqrt(d)).
    v16bf k00 = load_b_frag(Kbb, kH, jb,      0,  lane);
    v16bf k01 = load_b_frag(Kbb, kH, jb,      32, lane);
    v16bf k10 = load_b_frag(Kbb, kH, jb + 16, 0,  lane);
    v16bf k11 = load_b_frag(Kbb, kH, jb + 16, 32, lane);
    v8f s0 = {}, s1 = {};
    s0 = wmma_bf16(qf0, k00, s0);
    s0 = wmma_bf16(qf1, k01, s0);
    s1 = wmma_bf16(qf0, k10, s1);
    s1 = wmma_bf16(qf1, k11, s1);

    // Prefetch next key step while we do softmax (global_prefetch_b8).
    if (jb + 32 < jend) {
      __builtin_prefetch(Kbb + (size_t)(jb + 32 + lane) * kH, 0, 0);
      __builtin_prefetch(Vtb + (size_t)lane        * kT + jb + 32, 0, 0);
      __builtin_prefetch(Vtb + (size_t)(lane + 32) * kT + jb + 32, 0, 0);
    }

    // Causal mask + online softmax; row i of this lane = q0 + rbase + i.
    #pragma unroll
    for (int i = 0; i < 8; i++) {
      const int qrow = q0 + rbase + i;
      float a0 = ((jb      + nlo) <= qrow) ? s0[i] : -__builtin_inff();
      float a1 = ((jb + 16 + nlo) <= qrow) ? s1[i] : -__builtin_inff();
      float mt = fmaxf(a0, a1);
      #pragma unroll
      for (int d = 1; d < 16; d <<= 1) mt = fmaxf(mt, __shfl_xor(mt, d, 32));
      const float mnew  = fmaxf(m_i[i], mt);
      const float alpha = __expf(m_i[i] - mnew);
      const float p0 = __expf(a0 - mnew);
      const float p1 = __expf(a1 - mnew);
      float rs = p0 + p1;
      #pragma unroll
      for (int d = 1; d < 16; d <<= 1) rs += __shfl_xor(rs, d, 32);
      l_i[i] = l_i[i] * alpha + rs;
      m_i[i] = mnew;
      o0[i] *= alpha; o1[i] *= alpha; o2[i] *= alpha; o3[i] *= alpha;
      pbuf[wave][rbase + i][nlo]      = (bf16)p0;
      pbuf[wave][rbase + i][16 + nlo] = (bf16)p1;
    }

    // O += P * V  (V fragments already resident).
    const v16bf pf = load_a_frag(&pbuf[wave][0][0], 32, 0, 0, lane);
    o0 = wmma_bf16(pf, vf0, o0);
    o1 = wmma_bf16(pf, vf1, o1);
    o2 = wmma_bf16(pf, vf2, o2);
    o3 = wmma_bf16(pf, vf3, o3);
  }

  float* ob = out + (size_t)b * kT * kH;
  #pragma unroll
  for (int i = 0; i < 8; i++) {
    const float inv = 1.f / l_i[i];
    const size_t row = (size_t)(q0 + rbase + i) * kH;
    ob[row + 0  + nlo] = o0[i] * inv;
    ob[row + 16 + nlo] = o1[i] * inv;
    ob[row + 32 + nlo] = o2[i] * inv;
    ob[row + 48 + nlo] = o3[i] * inv;
  }
}

// ---------------------------------------------------------------------------
// ws layout: Wt bf16[3][64][512] @0 ; Qs @0x030000 ; Kb @0x430000 ;
// Vt (transposed) @0x830000. Total ~12.2 MiB.
// ---------------------------------------------------------------------------
extern "C" void kernel_launch(void* const* d_in, const int* in_sizes, int n_in,
                              void* d_out, int out_size, void* d_ws, size_t ws_size,
                              hipStream_t stream) {
  const float* x  = (const float*)d_in[0];
  const float* Wq = (const float*)d_in[1];
  const float* Wk = (const float*)d_in[2];
  const float* Wv = (const float*)d_in[3];
  float* out = (float*)d_out;

  char* ws = (char*)d_ws;
  bf16* Wt = (bf16*)(ws);
  bf16* Qs = (bf16*)(ws + 0x030000);
  bf16* Kb = (bf16*)(ws + 0x430000);
  bf16* Vt = (bf16*)(ws + 0x830000);

  wt_transpose_kernel<<<3, 256, 0, stream>>>(Wq, Wk, Wv, Wt);
  qkv_proj_kernel<<<kB * (kT / 16), 128, 0, stream>>>(x, Wt, Qs, Kb, Vt);
  attn_kernel<<<kB * (kT / 16) / 8, 256, 0, stream>>>(Qs, Kb, Vt, out);
}
